// GATEncoder_41575283425304
// MI455X (gfx1250) — matrix operand order repaired and checked by reading, not compiled, not run on hardware
//
#include <hip/hip_runtime.h>
#include <hip/hip_bf16.h>
#include <math.h>

#define N_NODES  10000
#define N_EDGES  160000
#define EF       (N_EDGES + N_NODES)   /* 170000 incl. self loops */
#define D        128
#define E_DIM    32
#define HEADS    3
#define CE       (D + E_DIM)           /* 160 */
#define NH       (HEADS * D)           /* 384 */
#define N_ETYPES 64
#define NEG_SLOPE 0.2f

typedef __bf16 bf16;
typedef __attribute__((ext_vector_type(16))) __bf16 v16bf;
typedef __attribute__((ext_vector_type(8)))  __bf16 v8bf;
typedef __attribute__((ext_vector_type(8)))  float  v8f;

__device__ __forceinline__ bf16 f2bf(float f) {
  union { float f; unsigned u; } v; v.f = f;
  unsigned r = v.u + 0x7FFFu + ((v.u >> 16) & 1u);   // round-to-nearest-even
  unsigned short hs = (unsigned short)(r >> 16);
  bf16 out;
  __builtin_memcpy(&out, &hs, sizeof(out));
  return out;
}

// monotone float <-> signed-int key for atomicMax on floats
__device__ __forceinline__ int f2ord(float f) {
  int i = __float_as_int(f);
  return i >= 0 ? i : (i ^ 0x7FFFFFFF);
}
__device__ __forceinline__ float ord2f(int k) {
  return __int_as_float(k >= 0 ? k : (k ^ 0x7FFFFFFF));
}

// ---------------------------------------------------------------------------
// utility fills / converts
// ---------------------------------------------------------------------------
__global__ void k_fill_u32(unsigned* __restrict__ p, unsigned v, long n) {
  long i = (long)blockIdx.x * blockDim.x + threadIdx.x;
  if (i < n) p[i] = v;
}

// convert + transpose: src row-major [K,N] f32 -> dst row-major [N,K] bf16
// (B operand of the WMMA GEMMs wants per-column-contiguous K)
__global__ void k_f32_to_bf16_T(const float* __restrict__ s, bf16* __restrict__ d,
                                int K, int N) {
  int tid = blockIdx.x * blockDim.x + threadIdx.x;
  if (tid >= K * N) return;
  int k = tid / N, n = tid - k * N;          // coalesced read of src
  d[(size_t)n * K + k] = f2bf(s[tid]);
}

// ---------------------------------------------------------------------------
// x = node_emb[x_idx] renormalized to L2 norm <= 1 (torch Embedding max_norm=1)
// one block (128 threads) per node; output bf16 feed for GEMM1
// ---------------------------------------------------------------------------
__global__ void k_norm_rows(const float* __restrict__ emb, const int* __restrict__ idx,
                            bf16* __restrict__ xb) {
  int n = blockIdx.x;
  int t = threadIdx.x;            // 0..127
  int row = idx[n];
  float v = emb[(size_t)row * D + t];
  float s = v * v;
  #pragma unroll
  for (int o = 16; o > 0; o >>= 1) s += __shfl_down(s, o);
  __shared__ float sh[4];
  if ((t & 31) == 0) sh[t >> 5] = s;
  __syncthreads();
  float tot = sh[0] + sh[1] + sh[2] + sh[3];
  float scale = 1.0f / fmaxf(sqrtf(tot), 1.0f);
  xb[(size_t)n * D + t] = f2bf(v * scale);
}

// ---------------------------------------------------------------------------
// bf16 WMMA GEMM: C[M,N] = A[M,K] * Bt^T   (+bias, optional ELU)
// A row-major [M,K] bf16; Bt row-major [N,K] bf16 (pre-transposed weights).
// One wave computes a 16x64 output tile: 1 A fragment feeds 4 WMMAs per K-step.
// K is a template param (multiple of 32) so the K-loop fully unrolls.
// A/B fragments: contiguous K (k = 16*half + s); C layout per CDNA5 ISA 7.12.2 (wave32).
// ---------------------------------------------------------------------------
template <int K>
__global__ void k_gemm_bf16(const bf16* __restrict__ A, const bf16* __restrict__ Bt,
                            float* __restrict__ Cf, bf16* __restrict__ Cbf,
                            const float* __restrict__ bias,
                            int M, int N, int act) {
  int wave = (int)(((long)blockIdx.x * blockDim.x + threadIdx.x) >> 5);
  int lane = threadIdx.x & 31;
  int tilesN = N >> 6;                       // 64-wide output tiles
  int tM = (wave / tilesN) << 4;
  int tN = (wave % tilesN) << 6;
  if (tM >= M) return;                       // wave-uniform guard

  int half = lane >> 4;                      // 0: lanes 0-15, 1: lanes 16-31
  int lm   = lane & 15;
  int kbA  = half << 4;                      // A: +16 K for upper half-wave
  int kbB  = half << 4;                      // B: +16 K for upper half-wave

  const bf16* arow = A + (size_t)(tM + lm) * K + kbA;
  const bf16* bcol[4];
  #pragma unroll
  for (int j = 0; j < 4; ++j)
    bcol[j] = Bt + (size_t)(tN + j * 16 + lm) * K + kbB;

  v8f acc[4] = {};
  #pragma unroll
  for (int k0 = 0; k0 < K; k0 += 32) {
    // A 16x32 fragment: lane holds 16 consecutive K (base..base+15), matching B
    v8bf alo = *(const v8bf*)(arow + k0);
    v8bf ahi = *(const v8bf*)(arow + k0 + 8);
    v16bf av;
    #pragma unroll
    for (int i = 0; i < 8; ++i) { av[i] = alo[i]; av[8 + i] = ahi[i]; }

    // B 32x16 fragments: lane holds 16 consecutive K of its column (contiguous now)
    #pragma unroll
    for (int j = 0; j < 4; ++j) {
      v16bf bv = *(const v16bf*)(bcol[j] + k0);
      acc[j] = __builtin_amdgcn_wmma_f32_16x16x32_bf16(false, av, false, bv,
                                                       (short)0, acc[j], false, false);
    }
  }

  // C layout: VGPR r holds M = r (lanes 0-15) / M = 8+r (lanes 16-31), N = lane&15
  int mrow = tM + (half << 3);
  #pragma unroll
  for (int j = 0; j < 4; ++j) {
    int n = tN + j * 16 + lm;
    float bn = bias ? bias[n] : 0.0f;
    #pragma unroll
    for (int r = 0; r < 8; ++r) {
      float v = acc[j][r] + bn;
      if (act == 1) v = v > 0.0f ? v : (expf(v) - 1.0f);   // ELU
      size_t off = (size_t)(mrow + r) * N + n;
      if (Cf)  Cf[off]  = v;
      if (Cbf) Cbf[off] = f2bf(v);
    }
  }
}

// ---------------------------------------------------------------------------
// per-node attention partial dots: ai[n,h]=<xh[n,h,:],att[h,0:128]>, aj likewise
// ---------------------------------------------------------------------------
__global__ void k_node_att(const float* __restrict__ xh, const float* __restrict__ att,
                           float* __restrict__ ai, float* __restrict__ aj) {
  int tid = blockIdx.x * blockDim.x + threadIdx.x;
  if (tid >= N_NODES * HEADS) return;
  int n = tid / HEADS, h = tid % HEADS;
  const float* xr = xh + (size_t)n * NH + h * D;
  const float* ar = att + h * (2 * D + E_DIM);
  float si = 0.f, sj = 0.f;
  for (int c = 0; c < D; ++c) { si += xr[c] * ar[c]; sj += xr[c] * ar[D + c]; }
  ai[tid] = si;
  aj[tid] = sj;
}

// per-edge-type attention term: aet[t,h] = <edge_emb[t,:], att[h,256:288]>
__global__ void k_aet(const float* __restrict__ edge_emb, const float* __restrict__ att,
                      float* __restrict__ aet) {
  int tid = blockIdx.x * blockDim.x + threadIdx.x;
  if (tid >= N_ETYPES * HEADS) return;
  int t = tid / HEADS, h = tid % HEADS;
  const float* ar = att + h * (2 * D + E_DIM) + 2 * D;
  float s = 0.f;
  for (int k = 0; k < E_DIM; ++k) s += edge_emb[t * E_DIM + k] * ar[k];
  aet[tid] = s;
}

// ---------------------------------------------------------------------------
// alpha[e,h] = leaky_relu(ai[dst]+aj[src]+aet[type]); running segment max on dst
// ---------------------------------------------------------------------------
__global__ void k_edge_alpha(const int* __restrict__ edge_index, const int* __restrict__ etype,
                             const float* __restrict__ ai, const float* __restrict__ aj,
                             const float* __restrict__ aet, float* __restrict__ alpha,
                             int* __restrict__ mord) {
  int tid = blockIdx.x * blockDim.x + threadIdx.x;
  if (tid >= HEADS * EF) return;
  int h = tid / EF;
  int e = tid - h * EF;
  int s, d2; float ae;
  if (e < N_EDGES) {
    s  = edge_index[e];
    d2 = edge_index[N_EDGES + e];
    ae = aet[etype[e] * HEADS + h];
  } else {
    s = d2 = e - N_EDGES;            // self loop, zero edge features
    ae = 0.f;
  }
  float v = ai[d2 * HEADS + h] + aj[s * HEADS + h] + ae;
  v = v > 0.f ? v : NEG_SLOPE * v;
  alpha[h * EF + e] = v;
  atomicMax(&mord[d2 * HEADS + h], f2ord(v));
}

// a = exp(alpha - m[dst]); denom[dst] += a   (in-place over alpha)
__global__ void k_edge_exp(const int* __restrict__ edge_index,
                           float* __restrict__ alpha, const int* __restrict__ mord,
                           float* __restrict__ denom) {
  int tid = blockIdx.x * blockDim.x + threadIdx.x;
  if (tid >= HEADS * EF) return;
  int h = tid / EF;
  int e = tid - h * EF;
  int d2 = (e < N_EDGES) ? edge_index[N_EDGES + e] : e - N_EDGES;
  float m = ord2f(mord[d2 * HEADS + h]);
  float v = __expf(alpha[h * EF + e] - m);
  alpha[h * EF + e] = v;
  atomicAdd(&denom[d2 * HEADS + h], v);
}

// ---------------------------------------------------------------------------
// weighted scatter aggregate: one wave per (edge, head); 160 channels / 32 lanes
// aggr[dst,h,c] += (a/denom) * cat(xh[src,h,:], edge_feat)[c]
// ---------------------------------------------------------------------------
__global__ void k_edge_aggr(const int* __restrict__ edge_index, const int* __restrict__ etype,
                            const float* __restrict__ a, const float* __restrict__ denom,
                            const float* __restrict__ xh, const float* __restrict__ edge_emb,
                            float* __restrict__ aggr) {
  int gw = (int)(((long)blockIdx.x * blockDim.x + threadIdx.x) >> 5);
  int lane = threadIdx.x & 31;
  if (gw >= HEADS * EF) return;
  int h = gw / EF;
  int e = gw - h * EF;
  int s, d2, t;
  if (e < N_EDGES) { s = edge_index[e]; d2 = edge_index[N_EDGES + e]; t = etype[e]; }
  else             { s = d2 = e - N_EDGES; t = -1; }
  float coef = a[h * EF + e] / denom[d2 * HEADS + h];
  float* out = aggr + (size_t)d2 * (HEADS * CE) + h * CE;
  const float* xs = xh + (size_t)s * NH + h * D;
  for (int c = lane; c < CE; c += 32) {
    float v = (c < D) ? xs[c] : ((t >= 0) ? edge_emb[t * E_DIM + (c - D)] : 0.f);
    atomicAdd(&out[c], coef * v);
  }
}

// z[n,c] = mean over heads of aggr[n,h,c], emitted as bf16 for GEMM2
__global__ void k_mean_heads(const float* __restrict__ aggr, bf16* __restrict__ zbf) {
  int tid = blockIdx.x * blockDim.x + threadIdx.x;
  if (tid >= N_NODES * CE) return;
  int n = tid / CE, c = tid - n * CE;
  const float* r = aggr + (size_t)n * (HEADS * CE);
  float v = (r[c] + r[CE + c] + r[2 * CE + c]) * (1.0f / 3.0f);
  zbf[tid] = f2bf(v);
}

// ---------------------------------------------------------------------------
static inline int cdiv(long a, int b) { return (int)((a + b - 1) / b); }

extern "C" void kernel_launch(void* const* d_in, const int* in_sizes, int n_in,
                              void* d_out, int out_size, void* d_ws, size_t ws_size,
                              hipStream_t stream) {
  (void)in_sizes; (void)n_in; (void)out_size; (void)ws_size;

  const int*   x_idx      = (const int*)d_in[0];
  const int*   edge_index = (const int*)d_in[1];
  const int*   etype      = (const int*)d_in[2];
  const float* node_emb   = (const float*)d_in[3];
  const float* edge_emb   = (const float*)d_in[4];
  const float* W[2]    = {(const float*)d_in[5],  (const float*)d_in[9]};
  const float* ATT[2]  = {(const float*)d_in[6],  (const float*)d_in[10]};
  const float* EUW[2]  = {(const float*)d_in[7],  (const float*)d_in[11]};
  const float* BIAS[2] = {(const float*)d_in[8],  (const float*)d_in[12]};

  char* p = (char*)d_ws;
  auto carve = [&](size_t bytes) -> void* {
    void* r = (void*)p;
    p += (bytes + 255) & ~(size_t)255;
    return r;
  };

  bf16*  xbf     = (bf16*)carve((size_t)N_NODES * D * 2);
  bf16*  wbf[2]  = {(bf16*)carve((size_t)D * NH * 2),  (bf16*)carve((size_t)D * NH * 2)};
  bf16*  ewbf[2] = {(bf16*)carve((size_t)CE * D * 2),  (bf16*)carve((size_t)CE * D * 2)};
  float* aet[2]  = {(float*)carve(N_ETYPES * HEADS * 4), (float*)carve(N_ETYPES * HEADS * 4)};
  float* xh    = (float*)carve((size_t)N_NODES * NH * 4);
  float* ai    = (float*)carve((size_t)N_NODES * HEADS * 4);
  float* aj    = (float*)carve((size_t)N_NODES * HEADS * 4);
  int*   mord  = (int*)  carve((size_t)N_NODES * HEADS * 4);
  float* denom = (float*)carve((size_t)N_NODES * HEADS * 4);
  float* alpha = (float*)carve((size_t)HEADS * EF * 4);
  float* aggr  = (float*)carve((size_t)N_NODES * HEADS * CE * 4);
  bf16*  zbf   = (bf16*) carve((size_t)N_NODES * CE * 2);

  // one-time weight conversions (transposed to [N,K] bf16) + per-layer aet tables
  for (int li = 0; li < 2; ++li) {
    k_f32_to_bf16_T<<<cdiv((long)D * NH, 256), 256, 0, stream>>>(W[li], wbf[li], D, NH);
    k_f32_to_bf16_T<<<cdiv((long)CE * D, 256), 256, 0, stream>>>(EUW[li], ewbf[li], CE, D);
    k_aet<<<1, 256, 0, stream>>>(edge_emb, ATT[li], aet[li]);
  }

  // x = maxnorm(node_emb[x_idx]) -> bf16
  k_norm_rows<<<N_NODES, 128, 0, stream>>>(node_emb, x_idx, xbf);

  const long nh3   = (long)N_NODES * HEADS;
  const long eh3   = (long)HEADS * EF;
  const long naggr = (long)N_NODES * HEADS * CE;

  for (int r = 0; r < 2; ++r) {
    for (int li = 0; li < 2; ++li) {
      // GEMM1: xh[N,384] = xbf[N,128] @ W[128,384]   (WMMA bf16, K=128)
      long waves1 = (long)(N_NODES / 16) * (NH / 64);
      k_gemm_bf16<D><<<cdiv(waves1 * 32, 256), 256, 0, stream>>>(
          xbf, wbf[li], xh, nullptr, nullptr, N_NODES, NH, 0);

      k_node_att<<<cdiv(nh3, 256), 256, 0, stream>>>(xh, ATT[li], ai, aj);

      k_fill_u32<<<cdiv(nh3, 256), 256, 0, stream>>>((unsigned*)mord, 0x807FFFFFu, nh3); // key(-inf)
      k_fill_u32<<<cdiv(nh3, 256), 256, 0, stream>>>((unsigned*)denom, 0u, nh3);

      k_edge_alpha<<<cdiv(eh3, 256), 256, 0, stream>>>(edge_index, etype, ai, aj,
                                                       aet[li], alpha, mord);
      k_edge_exp<<<cdiv(eh3, 256), 256, 0, stream>>>(edge_index, alpha, mord, denom);

      k_fill_u32<<<cdiv(naggr, 256), 256, 0, stream>>>((unsigned*)aggr, 0u, naggr);
      k_edge_aggr<<<cdiv(eh3 * 32, 256), 256, 0, stream>>>(edge_index, etype, alpha, denom,
                                                           xh, edge_emb, aggr);

      k_mean_heads<<<cdiv((long)N_NODES * CE, 256), 256, 0, stream>>>(aggr, zbf);

      // GEMM2: out[N,128] = z[N,160] @ euw[160,128] + b  (WMMA bf16, K=160; ELU on layer 0)
      bool last = (r == 1 && li == 1);
      long waves2 = (long)(N_NODES / 16) * (D / 64);
      k_gemm_bf16<CE><<<cdiv(waves2 * 32, 256), 256, 0, stream>>>(
          zbf, ewbf[li],
          last ? (float*)d_out : nullptr,
          last ? nullptr : xbf,
          BIAS[li], N_NODES, D, (li == 0) ? 1 : 0);
    }
  }
}
